// CoordinateLSTM_56495999811947
// MI455X (gfx1250) — matrix operand order repaired and checked by reading, not compiled, not run on hardware
//
#include <hip/hip_runtime.h>
#include <hip/hip_fp16.h>
#include <math.h>

typedef _Float16 v16h __attribute__((ext_vector_type(16)));
typedef _Float16 v2h  __attribute__((ext_vector_type(2)));
typedef float    v8f  __attribute__((ext_vector_type(8)));
typedef float    v2f  __attribute__((ext_vector_type(2)));

#define NROWS   32768          // B
#define KDIM    256            // I + H
#define HDIM    128            // H (per-gate output cols)
#define BM      64             // rows per m-block
#define MT      4              // 16-row m-tiles per block
#define KSTEPS  8              // 256 / 32
#define NGATES  4
#define FRAG_BYTES 1024        // 32 lanes * 32 bytes (16 halves/lane)
#define A_BYTES    (MT * KSTEPS * FRAG_BYTES)               // 32 KB
#define BG_BYTES   (8 * KSTEPS * FRAG_BYTES)                // 64 KB per gate
#define LDS_BYTES  (A_BYTES + NGATES * BG_BYTES)            // 288 KB (< 320 KB WGP LDS)
#define MBLOCKS    (NROWS / BM)                             // 512
#define NWG        256

// A-matrix 16x32 f16 fragment (ISA 7.12.2): lane = m + 16*(K-half),
// VGPR j<4: K = 8h + 2j + e ; j>=4: K = 16 + 8h + 2(j-4) + e.
__device__ __forceinline__ int a_off(int m, int k) {
    int kk = k & 31;
    int hh, j;
    if (kk < 16) { hh = kk >> 3; j = (kk & 7) >> 1; }
    else         { int s = kk - 16; hh = s >> 3; j = 4 + ((s & 7) >> 1); }
    return (m + (hh << 4)) * 32 + j * 4;
}

// B-matrix 32x16 f16 fragment (same K order as A): col n = lane%16, h = lane/16,
// VGPR v<4: K = 8h + 2v + e ; v>=4: K = 16 + 8h + 2(v-4) + e.
__device__ __forceinline__ int b_off(int n, int k) {
    int kk = k & 31;
    int lane = (n & 15) + (((kk >> 3) & 1) << 4);
    int v = ((kk >> 4) << 2) + ((kk & 7) >> 1);
    return lane * 32 + v * 4;
}

// gfx1250 hardware tanh (single TRANS op instead of libm exp-chain)
__device__ __forceinline__ float htanh(float x) {
#if __has_builtin(__builtin_amdgcn_tanhf)
    return __builtin_amdgcn_tanhf(x);
#else
    float r;
    asm volatile("v_tanh_f32 %0, %1\n\t"
                 "s_delay_alu instid0(TRANS32_DEP_1)"
                 : "=v"(r) : "v"(x));
    return r;
#endif
}
// sigmoid(x) = 0.5*tanh(x/2) + 0.5  -> one v_tanh_f32 + mul + fma
__device__ __forceinline__ float sigf(float t) {
    return __builtin_fmaf(htanh(t * 0.5f), 0.5f, 0.5f);
}

extern __shared__ __attribute__((aligned(256))) char smem[];

__global__ void __launch_bounds__(256)
coordlstm_wmma_kernel(const float* __restrict__ x,  const float* __restrict__ h,
                      const float* __restrict__ c,
                      const float* __restrict__ W1, const float* __restrict__ b1,
                      const float* __restrict__ W2, const float* __restrict__ b2,
                      const float* __restrict__ Wf, const float* __restrict__ bfp,
                      const float* __restrict__ W3, const float* __restrict__ b3,
                      float* __restrict__ out)
{
    const int tid  = threadIdx.x;
    const int wave = tid >> 5;          // n-tile id within each gate (0..7)
    const int lane = tid & 31;

    // ---- stage all gate weights into LDS once (f32 -> f16, B-fragment layout)
    const float* const Ws[NGATES] = {W1, W2, Wf, W3};
#pragma unroll
    for (int g = 0; g < NGATES; ++g) {
        const v2f* W = (const v2f*)Ws[g];
        char* gbase = smem + A_BYTES + g * BG_BYTES;
        for (int p = tid; p < HDIM * (KDIM / 2); p += 256) {
            int n = p >> 7;                  // weight row (= output col)
            int k = (p & 127) << 1;          // even K index
            v2f w = W[p];                    // row-major [128][256]
            v2h hv; hv.x = (_Float16)w.x; hv.y = (_Float16)w.y;
            char* frag = gbase + (((n >> 4) * KSTEPS + (k >> 5)) * FRAG_BYTES);
            *(v2h*)(frag + b_off(n, k)) = hv;
        }
    }

    // per-lane gate column and biases (fixed for the whole kernel)
    const int jcol = wave * 16 + (lane & 15);
    const float bias0 = b1[jcol], bias1 = b2[jcol], bias2 = bfp[jcol], bias3 = b3[jcol];
    const int hh = lane >> 4;

    __syncthreads();

    for (int mb = blockIdx.x; mb < MBLOCKS; mb += gridDim.x) {
        const int rowbase = mb * BM;

        // ---- stage A tile: concat(x,h) rows -> f16 A-fragments in LDS
        for (int p = tid; p < BM * (KDIM / 2); p += 256) {
            int rl = p >> 7;                 // local row 0..63
            int k  = (p & 127) << 1;         // even K index 0..254
            const v2f* src = (k < HDIM)
                ? (const v2f*)(x + (size_t)(rowbase + rl) * HDIM + k)
                : (const v2f*)(h + (size_t)(rowbase + rl) * HDIM + (k - HDIM));
            v2f a = __builtin_nontemporal_load(src);
            v2h hv; hv.x = (_Float16)a.x; hv.y = (_Float16)a.y;
            char* frag = smem + (((rl >> 4) * KSTEPS + (k >> 5)) * FRAG_BYTES);
            *(v2h*)(frag + a_off(rl & 15, k)) = hv;
        }
        __syncthreads();

        // speculative prefetch of next row-block (gfx1250 global_prefetch_b8)
        {
            int nmb = mb + gridDim.x;
            if (nmb < MBLOCKS && tid < BM) {
                __builtin_prefetch(x + (size_t)(nmb * BM + tid) * HDIM, 0, 1);
                __builtin_prefetch(h + (size_t)(nmb * BM + tid) * HDIM, 0, 1);
            }
        }

        // preload c for this block (overlaps global latency with WMMA loop)
        float cin[MT][8];
#pragma unroll
        for (int mt = 0; mt < MT; ++mt) {
            const int r0 = rowbase + mt * 16 + hh * 8;
#pragma unroll
            for (int v = 0; v < 8; ++v)
                cin[mt][v] = __builtin_nontemporal_load(
                    c + (size_t)(r0 + v) * HDIM + jcol);
        }

        // ---- main WMMA loop: 4 m-tiles x 4 gates, K = 8 x 32
        v8f acc[NGATES][MT];
        const v8f vzero = {0.f, 0.f, 0.f, 0.f, 0.f, 0.f, 0.f, 0.f};
#pragma unroll
        for (int g = 0; g < NGATES; ++g)
#pragma unroll
            for (int mt = 0; mt < MT; ++mt) acc[g][mt] = vzero;

        for (int ks = 0; ks < KSTEPS; ++ks) {
            v16h af[MT];
#pragma unroll
            for (int mt = 0; mt < MT; ++mt)
                af[mt] = *(const v16h*)(smem + (mt * KSTEPS + ks) * FRAG_BYTES + lane * 32);
#pragma unroll
            for (int g = 0; g < NGATES; ++g) {
                v16h bfrag = *(const v16h*)(smem + A_BYTES + g * BG_BYTES +
                                            (wave * KSTEPS + ks) * FRAG_BYTES + lane * 32);
#pragma unroll
                for (int mt = 0; mt < MT; ++mt)
                    acc[g][mt] = __builtin_amdgcn_wmma_f32_16x16x32_f16(
                        false, af[mt], false, bfrag, (short)0, acc[g][mt], false, false);
            }
        }
        __syncthreads();   // all LDS reads done before A is restaged

        // ---- fused LSTM epilogue: all 4 gate pre-activations resident per lane
        // C/D layout: col = lane%16 (jcol), row = rowbase + mt*16 + 8*(lane/16) + v
#pragma unroll
        for (int mt = 0; mt < MT; ++mt) {
            const int r0 = rowbase + mt * 16 + hh * 8;
#pragma unroll
            for (int v = 0; v < 8; ++v) {
                const size_t idx = (size_t)(r0 + v) * HDIM + jcol;
                float s1 = sigf(acc[0][mt][v] + bias0);    // forget gate
                float s2 = sigf(acc[1][mt][v] + bias1);    // input gate
                float fl = htanh(acc[2][mt][v] + bias2);   // candidate
                float s3 = sigf(acc[3][mt][v] + bias3);    // output gate
                float nc = cin[mt][v] * s1 + s2 * fl;
                float nh = htanh(nc) * s3;
                __builtin_nontemporal_store(nh, out + idx);                        // new_h
                __builtin_nontemporal_store(nc, out + (size_t)NROWS * HDIM + idx); // new_c
            }
        }
    }
}

extern "C" void kernel_launch(void* const* d_in, const int* in_sizes, int n_in,
                              void* d_out, int out_size, void* d_ws, size_t ws_size,
                              hipStream_t stream) {
    const float* x  = (const float*)d_in[0];
    const float* h  = (const float*)d_in[1];
    const float* c  = (const float*)d_in[2];
    const float* W1 = (const float*)d_in[3];
    const float* b1 = (const float*)d_in[4];
    const float* W2 = (const float*)d_in[5];
    const float* b2 = (const float*)d_in[6];
    const float* Wf = (const float*)d_in[7];
    const float* bf = (const float*)d_in[8];
    const float* W3 = (const float*)d_in[9];
    const float* b3 = (const float*)d_in[10];
    float* out = (float*)d_out;

    (void)hipFuncSetAttribute((const void*)coordlstm_wmma_kernel,
                              hipFuncAttributeMaxDynamicSharedMemorySize,
                              LDS_BYTES);
    coordlstm_wmma_kernel<<<dim3(NWG), dim3(256), LDS_BYTES, stream>>>(
        x, h, c, W1, b1, W2, b2, Wf, bf, W3, b3, out);
}
